// Conv_38225208934663
// MI455X (gfx1250) — matrix operand order repaired and checked by reading, not compiled, not run on hardware
//
#include <hip/hip_runtime.h>
#include <hip/hip_bf16.h>

typedef __attribute__((ext_vector_type(16))) __bf16       v16bf;
typedef __attribute__((ext_vector_type(8)))  float        v8f;
typedef __attribute__((ext_vector_type(4)))  unsigned int u32x4;

union Frag { u32x4 q[2]; v16bf v; };

#define ALPHA_NRM 0.17677669529663687f   /* 1/sqrt(32) */
#define INV_SQRT3 0.57735026918962576f

static __device__ __forceinline__ v8f wmma_bf16(v16bf a, v16bf b, v8f c) {
  // D = A(16x32 bf16) * B(32x16 bf16) + C(16x16 f32)
  return __builtin_amdgcn_wmma_f32_16x16x32_bf16(false, a, false, b, (short)0, c, false, false);
}

static __device__ __forceinline__ unsigned short bfhi(float v) {
  __bf16 b = (__bf16)v;
  return __builtin_bit_cast(unsigned short, b);
}
static __device__ __forceinline__ float bf2f(unsigned short s) {
  return (float)__builtin_bit_cast(__bf16, s);
}

// ---------------------------------------------------------------------------
// Prep: split w1/w2 into bf16 hi/lo pairs (bf16x3 emulated-f32 GEMM operands),
// and zero the accumulation buffers (graph-replay safe: runs every launch).
// ---------------------------------------------------------------------------
__global__ void __launch_bounds__(256)
prep_kernel(const float* __restrict__ w1, const float* __restrict__ w2,
            unsigned short* __restrict__ w1h, unsigned short* __restrict__ w1l,
            unsigned short* __restrict__ w2h, unsigned short* __restrict__ w2l,
            float* __restrict__ accb, float* __restrict__ cntb,
            int nAcc, int nCnt) {
  int t = blockIdx.x * 256 + threadIdx.x;
  if (t < 4096)  { float v = w1[t]; unsigned short h = bfhi(v); w1h[t] = h; w1l[t] = bfhi(v - bf2f(h)); }
  if (t < 65536) { float v = w2[t]; unsigned short h = bfhi(v); w2h[t] = h; w2l[t] = bfhi(v - bf2f(h)); }
  if (t < nAcc) accb[t] = 0.0f;
  if (t < nCnt) cntb[t] = 0.0f;
}

// ---------------------------------------------------------------------------
// Fused: gather -> MLP (2 WMMA GEMMs, bf16x3) -> e3nn tensor product consumed
// straight out of the WMMA accumulators -> atomic scatter-add.
// Block = 128 threads = 4 wave32 = 32 edges (2 edge-tiles of 16).
// Wave = TP path ip in {p00, p01, p10, p11}; processes BOTH edge-tiles so each
// w2 B-fragment load from L2 feeds two WMMA M-tiles.
// ---------------------------------------------------------------------------
__global__ void __launch_bounds__(128)
fused_kernel(const float* __restrict__ node_attr,
             const int*   __restrict__ edge_index,
             const float* __restrict__ edge_attr,
             const float* __restrict__ edge_sh,
             const float* __restrict__ b1,
             const float* __restrict__ b2,
             const unsigned short* __restrict__ w1h,
             const unsigned short* __restrict__ w1l,
             const unsigned short* __restrict__ w2h,
             const unsigned short* __restrict__ w2l,
             float* __restrict__ accOut,
             float* __restrict__ cntOut,
             int E) {
  __shared__ __align__(16) float  xbuf[32][64];
  __shared__ __align__(16) __bf16 eaHi[32][64];
  __shared__ __align__(16) __bf16 eaLo[32][64];
  __shared__ __align__(16) __bf16 hHi[32][64];
  __shared__ __align__(16) __bf16 hLo[32][64];
  __shared__ float a00s[32][16];   // x0 * sh0
  __shared__ float x0s [32][16];   // x0
  __shared__ float dts [32][16];   // <x1, sh1>/sqrt(3)
  __shared__ float c0s [32][16];   // x1[:, :, m]
  __shared__ float c1s [32][16];
  __shared__ float c2s [32][16];
  __shared__ float shb [32][4];
  __shared__ int   srcb[32];
  __shared__ int   dstb[32];
  __shared__ float resP00[2][16][16];
  __shared__ float resP11[2][16][16];
  __shared__ float resQ  [2][16][16];
  __shared__ float resC[3][2][16][16];

  const int tid = threadIdx.x;
  const int eb  = blockIdx.x * 32;

  if (tid < 32) {
    int e = eb + tid;
    bool ok = e < E;
    srcb[tid] = ok ? edge_index[e]     : 0;
    dstb[tid] = ok ? edge_index[E + e] : 0;
    #pragma unroll
    for (int m = 0; m < 4; m++) shb[tid][m] = ok ? edge_sh[e * 4 + m] : 0.0f;
  }
  __syncthreads();

  // Gather node rows + load/split edge_attr into bf16 hi/lo.
  for (int idx = tid; idx < 32 * 64; idx += 128) {
    int le = idx >> 6, c = idx & 63;
    int e = eb + le;
    bool ok = e < E;
    float xv = ok ? node_attr[srcb[le] * 64 + c] : 0.0f;
    xbuf[le][c] = xv;
    float ev = ok ? edge_attr[e * 64 + c] : 0.0f;
    unsigned short hb = bfhi(ev);
    eaHi[le][c] = __builtin_bit_cast(__bf16, hb);
    eaLo[le][c] = (__bf16)(ev - bf2f(hb));
  }
  __syncthreads();

  // Per-(edge,u) TP input scalars.
  for (int idx = tid; idx < 32 * 16; idx += 128) {
    int le = idx >> 4, u = idx & 15;
    float x0v = xbuf[le][u];
    float s0  = shb[le][0];
    a00s[le][u] = x0v * s0;
    x0s [le][u] = x0v;
    float x10 = xbuf[le][16 + u * 3 + 0];
    float x11 = xbuf[le][16 + u * 3 + 1];
    float x12 = xbuf[le][16 + u * 3 + 2];
    c0s[le][u] = x10; c1s[le][u] = x11; c2s[le][u] = x12;
    dts[le][u] = (x10 * shb[le][1] + x11 * shb[le][2] + x12 * shb[le][3]) * INV_SQRT3;
  }
  __syncthreads();

  const int lane = tid & 31;
  const int wv   = tid >> 5;    // 0..3
  const int ln   = lane & 15;
  const int hs   = lane >> 4;   // lane half: rows M = r + 8*hs

  // ---- GEMM1: h = silu(edge_attr @ w1 + b1) --------------------------------
  // Wave wv owns N-tile nt = wv for both edge-tiles; w1 fragments reused 2x.
  {
    const int nt = wv;
    v8f c0 = {0, 0, 0, 0, 0, 0, 0, 0};
    v8f c1 = {0, 0, 0, 0, 0, 0, 0, 0};
    #pragma unroll
    for (int ks = 0; ks < 2; ks++) {
      const int k0 = ks * 32;
      Frag Bh, Bl;
      #pragma unroll
      for (int s = 0; s < 16; s++) {
        const int kr = k0 + (s < 8 ? hs * 8 + s : 16 + hs * 8 + (s - 8));
        Bh.v[s] = __builtin_bit_cast(__bf16, w1h[kr * 64 + nt * 16 + ln]);
        Bl.v[s] = __builtin_bit_cast(__bf16, w1l[kr * 64 + nt * 16 + ln]);
      }
      #pragma unroll
      for (int et = 0; et < 2; et++) {
        const int er = et * 16 + ln;
        Frag Ah, Al;
        Ah.q[0] = *(const u32x4*)&eaHi[er][k0 + hs * 8];
        Ah.q[1] = *(const u32x4*)&eaHi[er][k0 + 16 + hs * 8];
        Al.q[0] = *(const u32x4*)&eaLo[er][k0 + hs * 8];
        Al.q[1] = *(const u32x4*)&eaLo[er][k0 + 16 + hs * 8];
        v8f acc = (et == 0) ? c0 : c1;
        acc = wmma_bf16(Ah.v, Bl.v, acc);
        acc = wmma_bf16(Al.v, Bh.v, acc);
        acc = wmma_bf16(Ah.v, Bh.v, acc);
        if (et == 0) c0 = acc; else c1 = acc;
      }
    }
    const float b1v = b1[nt * 16 + ln];
    #pragma unroll
    for (int et = 0; et < 2; et++) {
      v8f c = (et == 0) ? c0 : c1;
      #pragma unroll
      for (int r = 0; r < 8; r++) {
        float hv = c[r] + b1v;
        // silu via v_rcp_f32 (avoid IEEE divide expansion)
        float sv = hv * __builtin_amdgcn_rcpf(1.0f + __expf(-hv));
        const int er = et * 16 + r + hs * 8;
        unsigned short hb = bfhi(sv);
        hHi[er][nt * 16 + ln] = __builtin_bit_cast(__bf16, hb);
        hLo[er][nt * 16 + ln] = (__bf16)(sv - bf2f(hb));
      }
    }
  }
  __syncthreads();

  // ---- GEMM2 (tp_w = h @ w2 + b2) fused with TP u-contraction --------------
  {
    const int ip = wv;                 // TP path
    Frag Ah[2][2], Al[2][2];           // [et][ks]
    #pragma unroll
    for (int et = 0; et < 2; et++) {
      #pragma unroll
      for (int ks = 0; ks < 2; ks++) {
        const int k0 = ks * 32;
        const int er = et * 16 + ln;
        Ah[et][ks].q[0] = *(const u32x4*)&hHi[er][k0 + hs * 8];
        Ah[et][ks].q[1] = *(const u32x4*)&hHi[er][k0 + 16 + hs * 8];
        Al[et][ks].q[0] = *(const u32x4*)&hLo[er][k0 + hs * 8];
        Al[et][ks].q[1] = *(const u32x4*)&hLo[er][k0 + 16 + hs * 8];
      }
    }
    v8f acc0[2], acc1[2], acc2[2];
    #pragma unroll
    for (int et = 0; et < 2; et++) {
      acc0[et] = (v8f){0, 0, 0, 0, 0, 0, 0, 0};
      acc1[et] = (v8f){0, 0, 0, 0, 0, 0, 0, 0};
      acc2[et] = (v8f){0, 0, 0, 0, 0, 0, 0, 0};
    }
    for (int u = 0; u < 16; u++) {
      const int j0 = (ip * 16 + u) * 16;
      const float b2v = b2[j0 + ln];
      v8f d0 = {b2v, b2v, b2v, b2v, b2v, b2v, b2v, b2v};
      v8f d1 = d0;
      #pragma unroll
      for (int ks = 0; ks < 2; ks++) {
        Frag Bh, Bl;
        #pragma unroll
        for (int s = 0; s < 16; s++) {
          const int kr = ks * 32 + (s < 8 ? hs * 8 + s : 16 + hs * 8 + (s - 8));
          Bh.v[s] = __builtin_bit_cast(__bf16, w2h[kr * 1024 + j0 + ln]);
          Bl.v[s] = __builtin_bit_cast(__bf16, w2l[kr * 1024 + j0 + ln]);
        }
        d0 = wmma_bf16(Ah[0][ks].v, Bl.v, d0);
        d0 = wmma_bf16(Al[0][ks].v, Bh.v, d0);
        d0 = wmma_bf16(Ah[0][ks].v, Bh.v, d0);
        d1 = wmma_bf16(Ah[1][ks].v, Bl.v, d1);
        d1 = wmma_bf16(Al[1][ks].v, Bh.v, d1);
        d1 = wmma_bf16(Ah[1][ks].v, Bh.v, d1);
      }
      if (u < 15) {  // prefetch next u's w2 fragments into cache
        __builtin_prefetch((const void*)(w2h + (lane) * 1024 + j0 + 16), 0, 0);
        __builtin_prefetch((const void*)(w2l + (32 + lane) * 1024 + j0 + 16), 0, 0);
      }
      // D-tile lane layout: lane -> w = ln, VGPR r -> edge = et*16 + r + 8*hs.
      // Contract over u with per-edge scalar (LDS broadcast per lane-half).
      #pragma unroll
      for (int et = 0; et < 2; et++) {
        const v8f d = (et == 0) ? d0 : d1;
        #pragma unroll
        for (int r = 0; r < 8; r++) {
          const int le = et * 16 + r + hs * 8;
          const float dv = d[r];
          if      (ip == 0) acc0[et][r] += a00s[le][u] * dv;     // p00
          else if (ip == 1) acc0[et][r] += x0s [le][u] * dv;     // q01
          else if (ip == 3) acc0[et][r] += dts [le][u] * dv;     // p11
          else {                                                 // r10[m]
            acc0[et][r] += c0s[le][u] * dv;
            acc1[et][r] += c1s[le][u] * dv;
            acc2[et][r] += c2s[le][u] * dv;
          }
        }
      }
    }
    #pragma unroll
    for (int et = 0; et < 2; et++) {
      #pragma unroll
      for (int r = 0; r < 8; r++) {
        const int row = r + hs * 8;
        if      (ip == 0) resP00[et][row][ln] = acc0[et][r];
        else if (ip == 3) resP11[et][row][ln] = acc0[et][r];
        else if (ip == 1) resQ  [et][row][ln] = acc0[et][r];
        else {
          resC[0][et][row][ln] = acc0[et][r];
          resC[1][et][row][ln] = acc1[et][r];
          resC[2][et][row][ln] = acc2[et][r];
        }
      }
    }
  }
  __syncthreads();

  // ---- Finalize msg[e, 0:64] and scatter-add ------------------------------
  {
    const int le = tid >> 2;       // 0..31
    const int g  = tid & 3;
    const int e  = eb + le;
    if (e < E) {
      const int dst = dstb[le];
      const int et = le >> 4, row = le & 15;
      const float s0 = shb[le][0];
      #pragma unroll
      for (int q = 0; q < 16; q++) {
        const int ch = g * 16 + q;
        float val;
        if (ch < 16) {
          val = ALPHA_NRM * (resP00[et][row][ch] + resP11[et][row][ch]);
        } else {
          const int k = ch - 16;
          const int w_ = k / 3, m = k % 3;
          val = ALPHA_NRM * (resQ[et][row][w_] * shb[le][1 + m] + s0 * resC[m][et][row][w_]);
        }
        unsafeAtomicAdd(&accOut[dst * 64 + ch], val);
      }
      if (g == 0) unsafeAtomicAdd(&cntOut[dst], 1.0f);
    }
  }
}

// ---------------------------------------------------------------------------
// out = acc / max(cnt, 1)
// ---------------------------------------------------------------------------
__global__ void __launch_bounds__(256)
finish_kernel(const float* __restrict__ accb, const float* __restrict__ cntb,
              float* __restrict__ out, int nAcc) {
  int t = blockIdx.x * 256 + threadIdx.x;
  if (t < nAcc) out[t] = accb[t] / fmaxf(cntb[t >> 6], 1.0f);
}

extern "C" void kernel_launch(void* const* d_in, const int* in_sizes, int n_in,
                              void* d_out, int out_size, void* d_ws, size_t ws_size,
                              hipStream_t stream) {
  (void)n_in; (void)out_size; (void)ws_size;
  const float* node_attr  = (const float*)d_in[0];
  const int*   edge_index = (const int*)  d_in[1];
  const float* edge_attr  = (const float*)d_in[2];
  const float* edge_sh    = (const float*)d_in[3];
  const float* w1 = (const float*)d_in[4];
  const float* b1 = (const float*)d_in[5];
  const float* w2 = (const float*)d_in[6];
  const float* b2 = (const float*)d_in[7];

  const int N    = in_sizes[0] / 64;
  const int E    = in_sizes[3] / 4;
  const int nAcc = N * 64;

  char* ws = (char*)d_ws;
  float* accb = (float*)ws;
  float* cntb = (float*)(ws + (size_t)nAcc * 4);
  unsigned short* w1h = (unsigned short*)(ws + (size_t)(nAcc + N) * 4);
  unsigned short* w1l = w1h + 4096;
  unsigned short* w2h = w1l + 4096;
  unsigned short* w2l = w2h + 65536;

  const int prepTot = nAcc > 65536 ? nAcc : 65536;
  prep_kernel<<<(prepTot + 255) / 256, 256, 0, stream>>>(
      w1, w2, w1h, w1l, w2h, w2l, accb, cntb, nAcc, N);
  fused_kernel<<<(E + 31) / 32, 128, 0, stream>>>(
      node_attr, edge_index, edge_attr, edge_sh, b1, b2,
      w1h, w1l, w2h, w2l, accb, cntb, E);
  finish_kernel<<<(nAcc + 255) / 256, 256, 0, stream>>>(
      accb, cntb, (float*)d_out, nAcc);
}
